// MultiHeadAttention_65352222376751
// MI455X (gfx1250) — matrix operand order repaired and checked by reading, not compiled, not run on hardware
//
#include <hip/hip_runtime.h>

// ---------------------------------------------------------------------------
// CDNA5 (gfx1250) fused MHA: QKV proj -> causal flash attention -> out proj
// -> residual + LayerNorm.  Matrix math via v_wmma_f32_16x16x32_f16.
// K tiles staged by the Tensor Data Mover (TENSORcnt) when available,
// else async global->LDS (ASYNCcnt); GEMM A tiles async-staged.
// Wave32; 256-thread blocks = 8 waves.
// ---------------------------------------------------------------------------

typedef __attribute__((ext_vector_type(16))) _Float16 v16h;
typedef __attribute__((ext_vector_type(8)))  _Float16 v8h;
typedef __attribute__((ext_vector_type(8)))  float    v8f;

// ---- CDNA5 async global->LDS (GLOBAL_LOAD_ASYNC_TO_LDS_B128, ASYNCcnt) ----
#if __has_builtin(__builtin_amdgcn_global_load_async_to_lds_b128) && \
    __has_builtin(__builtin_amdgcn_s_wait_asynccnt)
#define USE_ASYNC_LDS 1
#else
#define USE_ASYNC_LDS 0
#endif

#if USE_ASYNC_LDS
// Builtin signature (probe-verified): param 0 is AS1 int4*, param 1 AS3 int4*.
typedef int v4i_vs __attribute__((vector_size(16)));
typedef __attribute__((address_space(1))) v4i_vs* global_b128_ptr;
typedef __attribute__((address_space(3))) v4i_vs* local_b128_ptr;

__device__ __forceinline__ void async_copy_b128(const void* gsrc, void* ldst) {
  __builtin_amdgcn_global_load_async_to_lds_b128(
      (global_b128_ptr)gsrc, (local_b128_ptr)ldst, 0, 0);
}
#endif

// ---- CDNA5 Tensor Data Mover (TENSOR_LOAD_TO_LDS, TENSORcnt) --------------
#if __has_builtin(__builtin_amdgcn_tensor_load_to_lds) && \
    __has_builtin(__builtin_amdgcn_s_wait_tensorcnt)
#define USE_TDM 1
typedef unsigned tdm_u32x4 __attribute__((vector_size(16)));
typedef int      tdm_i32x8 __attribute__((vector_size(32)));
typedef int      tdm_i32x4 __attribute__((vector_size(16)));
typedef __attribute__((address_space(3))) _Float16 lds_f16;
#else
#define USE_TDM 0
#endif

// ---- fragment loaders (layouts per CDNA5 ISA 7.12.2) ----------------------

// A fragment, 16x32 f16.  lane<16: row=lane, K=0..7 & 16..23;
// lane>=16: row=lane-16, K=8..15 & 24..31.  => two 16B chunks at koff,koff+16.
__device__ __forceinline__ v16h load_frag_a(const _Float16* base, int pitch) {
  const int lane = threadIdx.x & 31;
  const int row  = lane & 15;
  const int koff = (lane < 16) ? 0 : 8;
  const v8h c0 = *reinterpret_cast<const v8h*>(base + row * pitch + koff);
  const v8h c1 = *reinterpret_cast<const v8h*>(base + row * pitch + koff + 16);
  union { v16h v; v8h h[2]; } u;
  u.h[0] = c0; u.h[1] = c1;
  return u.v;
}

// B fragment, 32x16 f16, source stored N-major (row n holds K contiguously).
// lane<16: n=lane, K=0..7 & 16..23 ; lane>=16: n=lane-16, K=8..15 & 24..31.
__device__ __forceinline__ v16h load_frag_b(const _Float16* base, int pitch) {
  const int lane  = threadIdx.x & 31;
  const int n     = lane & 15;
  const int kbase = (lane < 16) ? 0 : 8;
  const v8h c0 = *reinterpret_cast<const v8h*>(base + n * pitch + kbase);
  const v8h c1 = *reinterpret_cast<const v8h*>(base + n * pitch + kbase + 16);
  union { v16h v; v8h h[2]; } u;
  u.h[0] = c0; u.h[1] = c1;
  return u.v;
}

__device__ __forceinline__ v8f wmma_f16(v16h a, v16h b, v8f c) {
  return __builtin_amdgcn_wmma_f32_16x16x32_f16(false, a, false, b,
                                                (short)0, c, false, false);
}

// ---------------------------------------------------------------------------
// GEMM: [8192 x 1024] x [1024 x 1024].  Block tile 128(M) x 128(N), K-step 32.
// Each wave: 32x64 = 2 A-frags x 4 B-frags = 8 WMMA per K-step.
// A_F16: A operand already f16 (ctx, async-staged) vs fp32 (converted).
// OUT_MODE 0: write f16 head-split [B,H,S,64];  1: write fp32 [M,1024].
// ---------------------------------------------------------------------------
template <bool A_F16, int OUT_MODE>
__global__ __launch_bounds__(256) void gemm1024(const void* __restrict__ Ap,
                                                const float* __restrict__ W,
                                                void* __restrict__ outp) {
  constexpr int Kd = 1024, Nd = 1024;
  __shared__ _Float16 As[128][32];   // M x K   (8 KB)
  __shared__ _Float16 Bs[128][32];   // N-major (8 KB)

  const int m0 = blockIdx.x * 128;
  const int n0 = blockIdx.y * 128;
  const int t = threadIdx.x, lane = t & 31, wave = t >> 5;
  const int wm = wave & 3;        // 0..3 -> 32-row M slab
  const int wn = wave >> 2;       // 0..1 -> 64-wide N slab
  const float*    Af = (const float*)Ap;
  const _Float16* Ah = (const _Float16*)Ap;

  v8f acc[2][4] = {};

  for (int k0 = 0; k0 < Kd; k0 += 32) {
    if (A_F16) {
#if USE_ASYNC_LDS
      // 128x32 f16 = 8 KB = 512 x 16B chunks; 2 async B128 copies per thread
#pragma unroll
      for (int i = 0; i < 2; ++i) {
        const int c = t + i * 256, r = c >> 2, p = c & 3;
        async_copy_b128(Ah + (size_t)(m0 + r) * Kd + k0 + p * 8,
                        &As[r][p * 8]);
      }
#else
#pragma unroll
      for (int i = 0; i < 16; ++i) {
        const int e = t + i * 256, r = e >> 5, c = e & 31;
        As[r][c] = Ah[(size_t)(m0 + r) * Kd + k0 + c];
      }
#endif
    } else {
#pragma unroll
      for (int i = 0; i < 16; ++i) {        // A tile 128x32 fp32 -> f16
        const int e = t + i * 256, r = e >> 5, c = e & 31;
        As[r][c] = (_Float16)Af[(size_t)(m0 + r) * Kd + k0 + c];
      }
    }
#pragma unroll
    for (int i = 0; i < 16; ++i) {          // B tile 32x128 -> N-major
      const int e = t + i * 256, k = e >> 7, n = e & 127;
      Bs[n][k] = (_Float16)W[(size_t)(k0 + k) * Nd + n0 + n];
    }
#if USE_ASYNC_LDS
    if (A_F16) __builtin_amdgcn_s_wait_asynccnt(0);
#endif
    __syncthreads();

    const v16h a0 = load_frag_a(&As[wm * 32][0], 32);
    const v16h a1 = load_frag_a(&As[wm * 32 + 16][0], 32);
#pragma unroll
    for (int j = 0; j < 4; ++j) {
      const v16h b = load_frag_b(&Bs[wn * 64 + j * 16][0], 32);
      acc[0][j] = wmma_f16(a0, b, acc[0][j]);
      acc[1][j] = wmma_f16(a1, b, acc[1][j]);
    }
    __syncthreads();
  }

  // epilogue: C fragment row = v + (lane<16?0:8), col = lane&15
#pragma unroll
  for (int sub = 0; sub < 2; ++sub) {
    const int rb = m0 + wm * 32 + sub * 16 + ((lane >> 4) << 3);
    const int cb = n0 + wn * 64 + (lane & 15);
#pragma unroll
    for (int j = 0; j < 4; ++j)
#pragma unroll
      for (int v = 0; v < 8; ++v) {
        const int m = rb + v;
        const int n = cb + j * 16;
        if (OUT_MODE == 0) {
          const int b = m >> 11, s = m & 2047;
          const int h = n >> 6,  d = n & 63;
          ((_Float16*)outp)[(((size_t)b * 16 + h) * 2048 + s) * 64 + d] =
              (_Float16)acc[sub][j][v];
        } else {
          ((float*)outp)[(size_t)m * Nd + n] = acc[sub][j][v];
        }
      }
  }
}

// ---------------------------------------------------------------------------
// Causal flash attention.  grid = (S/128, B*H).  Each wave owns 16 q-rows.
// K tiles staged by TDM (one D# per tile, wave 0, TENSORcnt ping-pong) or
// async-to-LDS; V transposed into LDS; P relayout through per-wave LDS.
// Qh/Kh/Vh: f16 [B*H, S, 64].  ctx: f16 [B, S, 1024].
// ---------------------------------------------------------------------------
__global__ __launch_bounds__(256) void attn_kernel(
    const _Float16* __restrict__ Qh, const _Float16* __restrict__ Kh,
    const _Float16* __restrict__ Vh, _Float16* __restrict__ ctx) {
  constexpr int S = 2048, D = 64;
  __shared__ _Float16 Ks[2][64][64];    // K tile ping-pong (16 KB)
  __shared__ _Float16 Vt[64][72];       // V^T tile: [d][key], padded pitch
  __shared__ _Float16 Ps[8][16][64];    // per-wave P staging (C->A relayout)

  const int qt = blockIdx.x;            // 0..15
  const int bh = blockIdx.y;            // 0..63
  const int b = bh >> 4, h = bh & 15;
  const int t = threadIdx.x, lane = t & 31, wave = t >> 5;
  const _Float16* Qb = Qh + (size_t)bh * S * D;
  const _Float16* Kb = Kh + (size_t)bh * S * D;
  const _Float16* Vb = Vh + (size_t)bh * S * D;

  const int qbase = qt * 128 + wave * 16;
  const int rbase = (lane >> 4) << 3;   // 0 or 8: C-frag row offset

  // Q fragments are loop-invariant: hoist (2 x 16x32 A fragments)
  const v16h aQ0 = load_frag_a(Qb + (size_t)qbase * D, D);
  const v16h aQ1 = load_frag_a(Qb + (size_t)qbase * D + 32, D);

  float mrow[8], lrow[8];
  v8f o[4] = {};
#pragma unroll
  for (int v = 0; v < 8; ++v) { mrow[v] = -1e30f; lrow[v] = 0.f; }

  const int nkt = 2 * qt + 2;           // causal: only tiles up to diagonal

#if USE_TDM
  // Stage 64x64 f16 K tile via Tensor Data Mover: one D# per tile, wave 0.
  // D# group 0: count=1 | lds_addr | global_addr[56:0] | type=2.
  // D# group 1: data_size=2B, tensor_dim0=64, tensor_dim1=2048,
  //             tile_dim0=64, tile_dim1=64, tensor_dim0_stride=64.
  auto stage_k = [&](int kt) {
    if (wave != 0) return;
    const int kc = kt * 64;
    const unsigned long long ga =
        (unsigned long long)(size_t)(Kb + (size_t)kc * D);
    const unsigned lds =
        (unsigned)(size_t)(lds_f16*)&Ks[kt & 1][0][0];
    tdm_u32x4 g0 = {1u, lds, (unsigned)ga,
                    (unsigned)(ga >> 32) | 0x80000000u};
    tdm_i32x8 g1 = {0x00010000, 64 << 16, 2048 << 16, 64 << 16, 64, 64, 0, 0};
    tdm_i32x4 gz = {0, 0, 0, 0};
#if defined(__clang_major__) && __clang_major__ >= 23
    tdm_i32x8 gz8 = {0, 0, 0, 0, 0, 0, 0, 0};
    __builtin_amdgcn_tensor_load_to_lds(g0, g1, gz, gz, gz8, 0);
#else
    __builtin_amdgcn_tensor_load_to_lds(g0, g1, gz, gz, 0);
#endif
  };
#else
  auto stage_k = [&](int kt) {          // stage 64x64 f16 K tile into LDS
    const int kc = kt * 64;
#if USE_ASYNC_LDS
    const int r = t >> 2, p = t & 3;    // 256 x 16B async chunks (4 KB)
    async_copy_b128(Kb + (size_t)(kc + r) * D + p * 8, &Ks[kt & 1][r][p * 8]);
#else
#pragma unroll
    for (int i = 0; i < 16; ++i) {
      const int e = t + i * 256, key = e >> 6, d = e & 63;
      Ks[kt & 1][key][d] = Kb[(size_t)(kc + key) * D + d];
    }
#endif
  };
#endif

  stage_k(0);                           // prologue: tile 0 in flight

  for (int kt = 0; kt < nkt; ++kt) {
    const int kc = kt * 64;
    if (kt + 1 < nkt) {
      stage_k(kt + 1);                  // prefetch next K tile (other buffer)
      __builtin_prefetch(Vb + (size_t)(kc + 64) * D + lane * 16, 0, 1);
    }

#pragma unroll
    for (int i = 0; i < 16; ++i) {      // cooperative transposed V load
      const int e = t + i * 256, key = e >> 6, d = e & 63;
      Vt[d][key] = Vb[(size_t)(kc + key) * D + d];
    }
#if USE_TDM
    if (wave == 0) {                    // ping-pong: oldest DMA must be done
      if (kt + 1 < nkt) __builtin_amdgcn_s_wait_tensorcnt(1);
      else              __builtin_amdgcn_s_wait_tensorcnt(0);
    }
#elif USE_ASYNC_LDS
    if (kt + 1 < nkt) __builtin_amdgcn_s_wait_asynccnt(1);
    else              __builtin_amdgcn_s_wait_asynccnt(0);
#endif
    __syncthreads();

    if (kc <= qbase + 15) {             // wave has visible keys in this tile
      const _Float16* Kt = &Ks[kt & 1][0][0];
      // ---- scores: 16 x 64 ----
      v8f sc[4] = {};
#pragma unroll
      for (int j = 0; j < 4; ++j) {
        const v16h bK0 = load_frag_b(Kt + (size_t)(j * 16) * 64, 64);
        const v16h bK1 = load_frag_b(Kt + (size_t)(j * 16) * 64 + 32, 64);
        sc[j] = wmma_f16(aQ0, bK0, sc[j]);
        sc[j] = wmma_f16(aQ1, bK1, sc[j]);
      }
      // ---- scale + causal mask ----
#pragma unroll
      for (int j = 0; j < 4; ++j)
#pragma unroll
        for (int v = 0; v < 8; ++v) {
          float x = sc[j][v] * 0.125f;  // 1/sqrt(64)
          const int row = qbase + rbase + v;
          const int col = kc + j * 16 + (lane & 15);
          if (col > row) x += -1e9f;
          sc[j][v] = x;
        }
      // ---- online softmax: row reductions across 16-lane halves ----
      float fsc[8], rsum[8];
#pragma unroll
      for (int v = 0; v < 8; ++v) {
        float x = fmaxf(fmaxf(sc[0][v], sc[1][v]), fmaxf(sc[2][v], sc[3][v]));
#pragma unroll
        for (int off = 1; off < 16; off <<= 1)
          x = fmaxf(x, __shfl_xor(x, off, 32));
        const float mnew = fmaxf(mrow[v], x);
        fsc[v] = __expf(mrow[v] - mnew);
        mrow[v] = mnew;
        rsum[v] = 0.f;
      }
#pragma unroll
      for (int j = 0; j < 4; ++j)
#pragma unroll
        for (int v = 0; v < 8; ++v) {
          const float p = __expf(sc[j][v] - mrow[v]);
          rsum[v] += p;
          Ps[wave][rbase + v][j * 16 + (lane & 15)] = (_Float16)p;
        }
#pragma unroll
      for (int v = 0; v < 8; ++v) {
#pragma unroll
        for (int off = 1; off < 16; off <<= 1)
          rsum[v] += __shfl_xor(rsum[v], off, 32);
        lrow[v] = lrow[v] * fsc[v] + rsum[v];
      }
#pragma unroll
      for (int j = 0; j < 4; ++j)       // rescale running O
#pragma unroll
        for (int v = 0; v < 8; ++v) o[j][v] *= fsc[v];

      // ---- O += P @ V ----
#pragma unroll
      for (int ks = 0; ks < 2; ++ks) {
        const v16h aP = load_frag_a(&Ps[wave][0][ks * 32], 64);
#pragma unroll
        for (int j = 0; j < 4; ++j) {
          const v16h bV = load_frag_b(&Vt[j * 16][0] + ks * 32, 72);
          o[j] = wmma_f16(aP, bV, o[j]);
        }
      }
    }
    __syncthreads();
  }

  // normalize + write ctx[b, s, h*64 + d] (f16)
#pragma unroll
  for (int j = 0; j < 4; ++j)
#pragma unroll
    for (int v = 0; v < 8; ++v) {
      const float val = o[j][v] / lrow[v];
      const int s = qbase + rbase + v;
      const int col = h * 64 + j * 16 + (lane & 15);
      ctx[((size_t)b * 2048 + s) * 1024 + col] = (_Float16)val;
    }
}

// ---------------------------------------------------------------------------
// Residual + LayerNorm: out = LN(Q + y).  One 256-thread block per row.
// ---------------------------------------------------------------------------
__global__ __launch_bounds__(256) void ln_kernel(
    const float* __restrict__ Qin, const float* __restrict__ y,
    const float* __restrict__ gamma, const float* __restrict__ beta,
    float* __restrict__ out) {
  const int row = blockIdx.x;
  const int t = threadIdx.x;
  __shared__ float red[16];

  float s = 0.f, s2 = 0.f, xv[4];
#pragma unroll
  for (int i = 0; i < 4; ++i) {
    const int c = t + i * 256;
    const float x = Qin[(size_t)row * 1024 + c] + y[(size_t)row * 1024 + c];
    xv[i] = x;
    s += x;
    s2 += x * x;
  }
#pragma unroll
  for (int off = 1; off < 32; off <<= 1) {
    s  += __shfl_xor(s, off, 32);
    s2 += __shfl_xor(s2, off, 32);
  }
  if ((t & 31) == 0) { red[t >> 5] = s; red[8 + (t >> 5)] = s2; }
  __syncthreads();
  s = 0.f; s2 = 0.f;
#pragma unroll
  for (int w = 0; w < 8; ++w) { s += red[w]; s2 += red[8 + w]; }
  const float mu  = s * (1.f / 1024.f);
  const float var = s2 * (1.f / 1024.f) - mu * mu;
  const float inv = rsqrtf(var + 1e-5f);
#pragma unroll
  for (int i = 0; i < 4; ++i) {
    const int c = t + i * 256;
    out[(size_t)row * 1024 + c] = (xv[i] - mu) * inv * gamma[c] + beta[c];
  }
}

// ---------------------------------------------------------------------------
extern "C" void kernel_launch(void* const* d_in, const int* in_sizes, int n_in,
                              void* d_out, int out_size, void* d_ws,
                              size_t ws_size, hipStream_t stream) {
  (void)in_sizes; (void)n_in; (void)out_size; (void)ws_size;
  const float* Q     = (const float*)d_in[0];
  const float* K     = (const float*)d_in[1];
  const float* V     = (const float*)d_in[2];
  const float* W_Q   = (const float*)d_in[3];
  const float* W_K   = (const float*)d_in[4];
  const float* W_V   = (const float*)d_in[5];
  const float* W_O   = (const float*)d_in[6];
  const float* gamma = (const float*)d_in[7];
  const float* beta  = (const float*)d_in[8];

  // workspace layout (96 MB total)
  char* ws = (char*)d_ws;
  _Float16* Qh  = (_Float16*)(ws);                          // 16 MB
  _Float16* Kh  = (_Float16*)(ws + (size_t)16 * (1 << 20));
  _Float16* Vh  = (_Float16*)(ws + (size_t)32 * (1 << 20));
  _Float16* ctx = (_Float16*)(ws + (size_t)48 * (1 << 20));
  float*    yb  = (float*)   (ws + (size_t)64 * (1 << 20)); // 32 MB

  const dim3 gG(64, 8);   // M/128, N/128
  gemm1024<false, 0><<<gG, 256, 0, stream>>>((const void*)Q, W_Q, (void*)Qh);
  gemm1024<false, 0><<<gG, 256, 0, stream>>>((const void*)K, W_K, (void*)Kh);
  gemm1024<false, 0><<<gG, 256, 0, stream>>>((const void*)V, W_V, (void*)Vh);

  attn_kernel<<<dim3(16, 64), 256, 0, stream>>>(Qh, Kh, Vh, ctx);

  gemm1024<true, 1><<<gG, 256, 0, stream>>>((const void*)ctx, W_O, (void*)yb);

  ln_kernel<<<8192, 256, 0, stream>>>(Q, yb, gamma, beta, (float*)d_out);
}
